// ScaledDotProductAttention_49246095016055
// MI455X (gfx1250) — compile-verified, matrix operand order checked by reading, not run on hardware
//
#include <hip/hip_runtime.h>
#include <hip/hip_bf16.h>
#include <stdint.h>

// B=2, H=16, S=2048, D=64; outputs: context [B,H,S,D] then attn [B,H,S,S] (f32)
typedef __attribute__((ext_vector_type(16))) __bf16 v16bf;
typedef __attribute__((ext_vector_type(4)))  __bf16 v4bf;
typedef __attribute__((ext_vector_type(8)))  float  v8f;

constexpr uint32_t kS  = 2048;
constexpr uint32_t kD  = 64;
constexpr uint32_t kE  = 4194304u;   // B*H*S*D elements per tensor
constexpr uint32_t kSD = 131072u;    // S*D
constexpr uint32_t kLdsBytes = 16u * kS * 2u + 16u * kD * 4u;  // 65536 + 4096

union FragAB { v16bf v; uint4 q[2]; };

// ---- pass 0: f32 -> bf16 (Q, K row-major; V transposed to [bh][d][s]) ----
__global__ __launch_bounds__(256)
void cvt_qkv_kernel(const float* __restrict__ Q, const float* __restrict__ K,
                    const float* __restrict__ V, __bf16* __restrict__ ws) {
  uint32_t i = blockIdx.x * 256u + threadIdx.x;
  if (i >= kE) return;
  ws[i]      = (__bf16)Q[i];
  ws[kE + i] = (__bf16)K[i];
  uint32_t bh  = i >> 17;            // / (S*D)
  uint32_t rem = i & (kSD - 1u);
  uint32_t s   = rem >> 6;
  uint32_t d   = rem & 63u;
  ws[2u * kE + bh * kSD + d * kS + s] = (__bf16)V[i];
}

// ---- main: one block = one (b,h, 16-row q tile); 8 waves ----
__global__ __launch_bounds__(256)
void fa_fwd_kernel(const __bf16* __restrict__ ws,
                   const uint8_t* __restrict__ mask,
                   float*         __restrict__ out) {
  extern __shared__ char smem_raw[];
  __bf16* Plds = (__bf16*)smem_raw;                       // 16 x 2048 bf16 = 64 KB
  float*  csc  = (float*)(smem_raw + 16u * kS * 2u);      // 4 x 16 x 16 f32 = 4 KB

  const uint32_t tid  = threadIdx.x;
  const uint32_t lane = tid & 31u;
  const uint32_t w    = tid >> 5;               // wave 0..7
  const uint32_t n    = lane & 15u;             // col (B/C) or row M (A) index
  const uint32_t kg   = lane >> 4;              // half-wave group

  const uint32_t bid   = blockIdx.x;            // 0..4095
  const uint32_t bh    = bid >> 7;
  const uint32_t qt    = bid & 127u;
  const uint32_t qbase = qt << 4;
  const uint32_t bI    = bh >> 4;               // batch index

  const __bf16* Qbf = ws;
  const __bf16* Kbf = ws + kE;
  const __bf16* VTb = ws + 2u * kE;
  const uint32_t tens = bh * kSD;

  // ---- A fragments: Q rows qbase..qbase+15, K-dim halves [0,32) and [32,64) ----
  FragAB a0, a1;
  {
    const uint4* p = (const uint4*)(Qbf + tens + (qbase + n) * kD);
    a0.q[0] = p[kg];      a0.q[1] = p[2u + kg];   // d = 8*kg, 16+8*kg
    a1.q[0] = p[4u + kg]; a1.q[1] = p[6u + kg];   // d = 32+8*kg, 48+8*kg
  }

  // ---- GEMM1: this wave's 256 keys -> 16 score tiles (scaled) into LDS ----
  for (uint32_t t = 0; t < 16u; ++t) {
    const uint32_t kb = w * 256u + t * 16u;
    if (t < 15u)  // prefetch next K tile (global_prefetch_b8)
      __builtin_prefetch(Kbf + tens + (kb + 16u + n) * kD, 0, 1);

    FragAB b0, b1;
    const uint4* p = (const uint4*)(Kbf + tens + (kb + n) * kD);
    b0.q[0] = p[kg];      b0.q[1] = p[2u + kg];
    b1.q[0] = p[4u + kg]; b1.q[1] = p[6u + kg];

    v8f c = {0.f, 0.f, 0.f, 0.f, 0.f, 0.f, 0.f, 0.f};
    c = __builtin_amdgcn_wmma_f32_16x16x32_bf16(false, a0.v, false, b0.v,
                                                (short)0, c, false, false);
    c = __builtin_amdgcn_wmma_f32_16x16x32_bf16(false, a1.v, false, b1.v,
                                                (short)0, c, false, false);

    // C layout: lane -> col n, half kg; VGPR r -> row (r + 8*kg)
#pragma unroll
    for (uint32_t r = 0; r < 8u; ++r)
      Plds[(r + 8u * kg) * kS + kb + n] = (__bf16)(c[r] * 0.125f); // 1/sqrt(64)
  }
  __syncthreads();

  // ---- softmax (+mask): wave w owns rows 2w, 2w+1; 4-wide vector scans ----
#pragma unroll
  for (uint32_t rr = 0; rr < 2u; ++rr) {
    const uint32_t row = w * 2u + rr;
    __bf16* prow = &Plds[row * kS];
    const uint8_t* mrow = mask + (size_t)bI * kS * kS + (size_t)(qbase + row) * kS;

    // pass A: apply mask, row max   (ds_load_b64 / ds_store_b64, b32 mask loads)
    float mx = -3.4e38f;
    for (uint32_t i = 0; i < 16u; ++i) {
      const uint32_t c0 = i * 128u + 4u * lane;
      v4bf p4 = *(const v4bf*)(prow + c0);
      const uint32_t mb = *(const uint32_t*)(mrow + c0);
#pragma unroll
      for (uint32_t j = 0; j < 4u; ++j) {
        float x = (float)p4[j];
        if ((mb >> (8u * j)) & 0xFFu) x = -1e9f;     // mask==true -> NEG_BIG
        p4[j] = (__bf16)x;
        mx = fmaxf(mx, x);
      }
      *(v4bf*)(prow + c0) = p4;
    }
#pragma unroll
    for (int o = 16; o > 0; o >>= 1)
      mx = fmaxf(mx, __shfl_xor(mx, o, 32));

    // pass B: exp(x - max), partial sum
    float sum = 0.f;
    for (uint32_t i = 0; i < 16u; ++i) {
      const uint32_t c0 = i * 128u + 4u * lane;
      v4bf p4 = *(const v4bf*)(prow + c0);
#pragma unroll
      for (uint32_t j = 0; j < 4u; ++j) {
        float e = __expf((float)p4[j] - mx);
        p4[j] = (__bf16)e;
        sum += e;
      }
      *(v4bf*)(prow + c0) = p4;
    }
#pragma unroll
    for (int o = 16; o > 0; o >>= 1)
      sum += __shfl_xor(sum, o, 32);
    const float inv = 1.0f / sum;

    // pass C: normalize; attn out as float4 (global_store_b128), bf16 P back
    float* attn = out + kE + (size_t)bh * kS * kS + (size_t)(qbase + row) * kS;
    for (uint32_t i = 0; i < 16u; ++i) {
      const uint32_t c0 = i * 128u + 4u * lane;
      v4bf p4 = *(const v4bf*)(prow + c0);
      float4 f4;
      f4.x = (float)p4[0] * inv;  f4.y = (float)p4[1] * inv;
      f4.z = (float)p4[2] * inv;  f4.w = (float)p4[3] * inv;
      *(float4*)(attn + c0) = f4;
      p4[0] = (__bf16)f4.x; p4[1] = (__bf16)f4.y;
      p4[2] = (__bf16)f4.z; p4[3] = (__bf16)f4.w;
      *(v4bf*)(prow + c0) = p4;
    }
  }
  __syncthreads();

  // ---- GEMM2: context = P(16x2048) @ V(2048x64); 8 waves = 4 col-tiles x 2 key halves
  const uint32_t nt   = w & 3u;                 // output col tile (16 cols)
  const uint32_t half = w >> 2;                 // key half: [0,1024) or [1024,2048)
  v8f acc = {0.f, 0.f, 0.f, 0.f, 0.f, 0.f, 0.f, 0.f};
  {
    const uint4* vp = (const uint4*)(VTb + tens + (nt * 16u + n) * kS); // V^T row = d
    const uint4* ap = (const uint4*)(Plds + n * kS);                    // A row m = n
    const uint32_t k0 = half * 1024u;
    for (uint32_t kc = k0; kc < k0 + 1024u; kc += 32u) {
      FragAB af, bfr;
      af.q[0]  = ap[(kc + 8u * kg) >> 3];
      af.q[1]  = ap[(kc + 16u + 8u * kg) >> 3];
      bfr.q[0] = vp[(kc + 8u * kg) >> 3];
      bfr.q[1] = vp[(kc + 16u + 8u * kg) >> 3];
      acc = __builtin_amdgcn_wmma_f32_16x16x32_bf16(false, af.v, false, bfr.v,
                                                    (short)0, acc, false, false);
    }
  }
  if (w >= 4u) {                                 // upper waves deposit partials
#pragma unroll
    for (uint32_t r = 0; r < 8u; ++r)
      csc[nt * 256u + (r + 8u * kg) * 16u + n] = acc[r];
  }
  __syncthreads();
  if (w < 4u) {                                  // lower waves merge + store
    float* ctx = out + (size_t)tens + (size_t)qbase * kD;
#pragma unroll
    for (uint32_t r = 0; r < 8u; ++r) {
      float v = acc[r] + csc[nt * 256u + (r + 8u * kg) * 16u + n];
      ctx[(size_t)(r + 8u * kg) * kD + nt * 16u + n] = v;
    }
  }
}

extern "C" void kernel_launch(void* const* d_in, const int* in_sizes, int n_in,
                              void* d_out, int out_size, void* d_ws, size_t ws_size,
                              hipStream_t stream) {
  (void)in_sizes; (void)n_in; (void)out_size; (void)ws_size;
  const float*   Q    = (const float*)d_in[0];
  const float*   K    = (const float*)d_in[1];
  const float*   V    = (const float*)d_in[2];
  const uint8_t* mask = (const uint8_t*)d_in[3];   // jnp.bool_ = 1 byte/elem
  __bf16* ws  = (__bf16*)d_ws;                     // 3 bf16 tensors = 25.2 MB
  float*  out = (float*)d_out;

  // allow 68 KB dynamic LDS (idempotent; not a stream op, graph-capture safe)
  (void)hipFuncSetAttribute((const void*)fa_fwd_kernel,
                            hipFuncAttributeMaxDynamicSharedMemorySize,
                            (int)kLdsBytes);

  cvt_qkv_kernel<<<kE / 256u, 256, 0, stream>>>(Q, K, V, ws);
  fa_fwd_kernel<<<4096, 256, kLdsBytes, stream>>>(ws, mask, out);
}
